// PerlinAttention_23106924053051
// MI455X (gfx1250) — matrix operand order repaired and checked by reading, not compiled, not run on hardware
//
#include <hip/hip_runtime.h>
#include <hip/hip_bf16.h>
#include <math.h>

typedef __attribute__((ext_vector_type(16))) _Float16 v16h;
typedef __attribute__((ext_vector_type(8)))  _Float16 v8h;
typedef __attribute__((ext_vector_type(8)))  float    v8f;

#define DEVI __device__ __forceinline__

namespace pk {

// ---------------- problem constants ----------------
constexpr int Bc = 4, Hc = 16, Tc = 1024, Dc = 64;
constexpr int BH = Bc * Hc;                // 64 batched heads
constexpr int MF = 266, MP = 288;          // performer features (padded to mult of 32)
constexpr int D3 = 192, D2 = 128;          // 3*D, 2*D
constexpr int W2 = 512;                    // 2*L
constexpr float DN         = 0.35355339059327373f;  // 64^-0.25
constexpr float INV_SQRT_M = 0.06131393f;           // 1/sqrt(266)

// ---------------- workspace layout (bytes) ----------------
constexpr size_t A256(size_t x) { return (x + 255) & ~(size_t)255; }
constexpr size_t OFF_Q16   = 0;
constexpr size_t OFF_K16   = A256(OFF_Q16   + (size_t)BH*Tc*Dc*2);
constexpr size_t OFF_DIAGQ = A256(OFF_K16   + (size_t)BH*Tc*Dc*2);
constexpr size_t OFF_DIAGK = A256(OFF_DIAGQ + (size_t)BH*Tc*4);
constexpr size_t OFF_ZINV  = A256(OFF_DIAGK + (size_t)BH*Tc*4);
constexpr size_t OFF_SCALE = A256(OFF_ZINV  + (size_t)BH*Tc*4);
constexpr size_t OFF_THR   = A256(OFF_SCALE + (size_t)BH*Tc*4);
constexpr size_t OFF_KRMAX = A256(OFF_THR   + (size_t)BH*Tc*4);
constexpr size_t OFF_KSTAB = A256(OFF_KRMAX + (size_t)BH*Tc*4);
constexpr size_t OFF_KFSUM = A256(OFF_KSTAB + (size_t)BH*4);
constexpr size_t OFF_WP16  = A256(OFF_KFSUM + (size_t)BH*MP*4);
constexpr size_t OFF_WENCT = A256(OFF_WP16  + (size_t)MP*Dc*2);
constexpr size_t OFF_WDECT = A256(OFF_WENCT + (size_t)D2*D3*2);
constexpr size_t OFF_V3T   = A256(OFF_WDECT + (size_t)W2*D2*2);
constexpr size_t OFF_QF    = A256(OFF_V3T   + (size_t)BH*D3*Tc*2);
constexpr size_t OFF_KFT   = A256(OFF_QF    + (size_t)BH*Tc*MP*2);
constexpr size_t OFF_KVT   = A256(OFF_KFT   + (size_t)BH*MP*Tc*2);
constexpr size_t OFF_TPERF = A256(OFF_KVT   + (size_t)BH*D3*MP*2);
constexpr size_t OFF_ENC   = A256(OFF_TPERF + (size_t)BH*Tc*D3*2);
constexpr size_t OFF_H1    = A256(OFF_ENC   + (size_t)BH*Tc*D2*4);
constexpr size_t OFF_ROWC  = A256(OFF_H1    + (size_t)BH*Tc*D2*2);
constexpr size_t OFF_PROJQ = A256(OFF_ROWC  + (size_t)Bc*64*Tc*128*2);
constexpr size_t OFF_PROJK = A256(OFF_PROJQ + (size_t)BH*Tc*MP*4);
constexpr size_t OFF_CTX   = A256(OFF_PROJK + (size_t)BH*Tc*MP*4);
constexpr size_t WS_NEED   = OFF_CTX + (size_t)BH*Tc*Dc*4;
// aliases into the dead proj_q/proj_k region (lifetimes disjoint, exact fit):
constexpr size_t OFF_Y1    = OFF_PROJQ;                          // f16 [4,64,512,128]
constexpr size_t OFF_Y2    = OFF_Y1 + (size_t)Bc*64*512*128*2;   // f16 [4,64,512,128]
constexpr size_t OFF_Y3    = OFF_Y2 + (size_t)Bc*64*512*128*2;   // f16 [4,16,1024,128]
constexpr size_t OFF_PROBS = OFF_Y3 + (size_t)Bc*16*1024*128*2;  // f32 [4,16,1024,256]

// ---------------- wave32 reductions ----------------
DEVI float wsum(float v) {
#pragma unroll
  for (int m = 16; m > 0; m >>= 1) v += __shfl_xor(v, m, 32);
  return v;
}
DEVI float wmaxr(float v) {
#pragma unroll
  for (int m = 16; m > 0; m >>= 1) v = fmaxf(v, __shfl_xor(v, m, 32));
  return v;
}
DEVI void lds_fence() {  // intra-wave LDS producer->consumer ordering (CDNA5 split counters)
  asm volatile("s_wait_dscnt 0x0" ::: "memory");
}

// ---------------- WMMA fragment loads (per ISA 7.12.2 layouts) ----------------
// A 16x32 f16, row-major source: lane = m%16 | (K-half), two 16B loads
DEVI v16h load_a_frag(const _Float16* __restrict__ A, int lda, int k0, int lane) {
  const _Float16* p = A + (size_t)(lane & 15) * lda + k0 + ((lane >> 4) << 3);
  v8h lo = *(const v8h*)p;
  v8h hi = *(const v8h*)(p + 16);
  return __builtin_shufflevector(lo, hi, 0,1,2,3,4,5,6,7,8,9,10,11,12,13,14,15);
}
// B 32x16 f16 given as Bt row-major [N,K]: lane = n | (K-half), two 16B loads (same K order as A)
DEVI v16h load_b_frag(const _Float16* __restrict__ Bt, int ldb, int k0, int lane) {
  const _Float16* p = Bt + (size_t)(lane & 15) * ldb + k0 + ((lane >> 4) << 3);
  v8h lo = *(const v8h*)p;
  v8h hi = *(const v8h*)(p + 16);
  return __builtin_shufflevector(lo, hi, 0,1,2,3,4,5,6,7,8,9,10,11,12,13,14,15);
}

template <int NT>
DEVI void gemm_core(const _Float16* __restrict__ A, int lda,
                    const _Float16* __restrict__ Bt, int ldb,
                    int K, int lane, v8f (&acc)[NT]) {
  for (int k0 = 0; k0 < K; k0 += 32) {
    v16h a = load_a_frag(A, lda, k0, lane);
#pragma unroll
    for (int i = 0; i < NT; ++i) {
      v16h b = load_b_frag(Bt + (size_t)(i * 16) * ldb, ldb, k0, lane);
      acc[i] = __builtin_amdgcn_wmma_f32_16x16x32_f16(false, a, false, b, (short)0,
                                                      acc[i], false, false);
    }
  }
}

// ================= preparation kernels =================
__global__ __launch_bounds__(256) void k_prep_weights(
    const float* __restrict__ Wperf, const float* __restrict__ Wenc,
    const float* __restrict__ Wdec, _Float16* __restrict__ wp,
    _Float16* __restrict__ wencT, _Float16* __restrict__ wdecT) {
  int idx = blockIdx.x * 256 + threadIdx.x;
  if (idx < MP * Dc) {                       // W_perf (padded rows -> 0), acts as Bt
    int m = idx / Dc, d = idx % Dc;
    wp[idx] = (m < MF) ? (_Float16)Wperf[m * Dc + d] : (_Float16)0.f;
  } else if (idx < MP * Dc + D2 * D3) {      // W_enc^T  [128,192]
    int i = idx - MP * Dc;
    int n = i / D3, kk = i % D3;
    wencT[i] = (_Float16)Wenc[kk * D2 + n];
  } else if (idx < MP * Dc + D2 * D3 + W2 * D2) {  // W_dec^T [512,128]
    int i = idx - MP * Dc - D2 * D3;
    int n = i / D2, kk = i % D2;
    wdecT[i] = (_Float16)Wdec[kk * W2 + n];
  }
}

// q16/k16 = x * D^-0.25 (f16), diag = 0.5*||x*dn||^2
__global__ __launch_bounds__(256) void k_scale_convert(
    const float* __restrict__ q, const float* __restrict__ k,
    _Float16* __restrict__ q16, _Float16* __restrict__ k16,
    float* __restrict__ dq, float* __restrict__ dk) {
  int wid = (blockIdx.x * 256 + threadIdx.x) >> 5, lane = threadIdx.x & 31;
  int sel = wid >> 16;       // 0: q rows, 1: k rows (65536 each)
  int row = wid & 65535;
  const float* src = sel ? k : q;
  _Float16* dst = sel ? k16 : q16;
  float* dd = sel ? dk : dq;
  float x0 = src[(size_t)row * 64 + lane] * DN;
  float x1 = src[(size_t)row * 64 + lane + 32] * DN;
  dst[(size_t)row * 64 + lane] = (_Float16)x0;
  dst[(size_t)row * 64 + lane + 32] = (_Float16)x1;
  float s = wsum(x0 * x0 + x1 * x1);
  if (lane == 0) dd[row] = 0.5f * s;
}

// v3^T [bh][d][t]: d<64 -> v, d>=64 -> nearest-resized identity (t == 8*(d-64)+4)
__global__ __launch_bounds__(256) void k_build_v3T(const float* __restrict__ v,
                                                   _Float16* __restrict__ v3t) {
  int idx = blockIdx.x * 256 + threadIdx.x;  // BH*192*1024
  int t = idx & 1023;
  int d = (idx >> 10) % D3;
  int bh = idx / (D3 * 1024);
  float val;
  if (d < 64) val = v[((size_t)bh * 1024 + t) * 64 + d];
  else        val = (t == ((d - 64) * 8 + 4)) ? 1.f : 0.f;
  v3t[idx] = (_Float16)val;
}

// ================= GEMM kernels =================
// proj = (x*dn) @ W_perf^T : [1024,64]x[64,288] per head
__global__ __launch_bounds__(256) void k_gemm_proj(const _Float16* __restrict__ X16,
                                                   const _Float16* __restrict__ Wp,
                                                   float* __restrict__ proj) {
  int gw = (blockIdx.x * 256 + threadIdx.x) >> 5, lane = threadIdx.x & 31;
  const int WPB = 64 * 6;
  int bh = gw / WPB, r = gw % WPB;
  int m0 = (r / 6) * 16, n0 = (r % 6) * 48;
  const _Float16* A = X16 + (size_t)bh * Tc * Dc + (size_t)m0 * Dc;
  const _Float16* Bt = Wp + (size_t)n0 * Dc;
  v8f acc[3] = {};
  gemm_core<3>(A, Dc, Bt, Dc, Dc, lane, acc);
  float* Dp = proj + (size_t)bh * Tc * MP;
  int rr = (lane >> 4) << 3, c0 = lane & 15;
#pragma unroll
  for (int i = 0; i < 3; ++i)
#pragma unroll
    for (int r8 = 0; r8 < 8; ++r8)
      Dp[(size_t)(m0 + rr + r8) * MP + (n0 + i * 16 + c0)] = acc[i][r8];
}

// qf = exp(proj - diag - rowmax)/sqrt(M) + 1e-6  (padded features -> 0)
__global__ __launch_bounds__(256) void k_qf(const float* __restrict__ proj,
                                            const float* __restrict__ diag,
                                            _Float16* __restrict__ qf) {
  int row = (blockIdx.x * 256 + threadIdx.x) >> 5, lane = threadIdx.x & 31;
  const float* pr = proj + (size_t)row * MP;
  float mx = -3.4e38f;
  for (int m = lane; m < MF; m += 32) mx = fmaxf(mx, pr[m]);
  mx = wmaxr(mx);
  float dg = diag[row];
  for (int m = lane; m < MP; m += 32) {
    float val = 0.f;
    if (m < MF) val = expf(pr[m] - dg - mx) * INV_SQRT_M + 1e-6f;
    qf[(size_t)row * MP + m] = (_Float16)val;
  }
}

__global__ __launch_bounds__(256) void k_krowmax(const float* __restrict__ proj,
                                                 float* __restrict__ rmax) {
  int row = (blockIdx.x * 256 + threadIdx.x) >> 5, lane = threadIdx.x & 31;
  const float* pr = proj + (size_t)row * MP;
  float mx = -3.4e38f;
  for (int m = lane; m < MF; m += 32) mx = fmaxf(mx, pr[m]);
  mx = wmaxr(mx);
  if (lane == 0) rmax[row] = mx;
}

__global__ __launch_bounds__(256) void k_kstab(const float* __restrict__ rmax,
                                               float* __restrict__ stab) {
  int wid = (blockIdx.x * 256 + threadIdx.x) >> 5, lane = threadIdx.x & 31;  // wid < 64
  float mx = -3.4e38f;
  for (int t = lane; t < 1024; t += 32) mx = fmaxf(mx, rmax[(size_t)wid * 1024 + t]);
  mx = wmaxr(mx);
  if (lane == 0) stab[wid] = mx;
}

// kf^T [bh][m][t]  (global stab per head)
__global__ __launch_bounds__(256) void k_kf(const float* __restrict__ projk,
                                            const float* __restrict__ diagk,
                                            const float* __restrict__ stab,
                                            _Float16* __restrict__ kft) {
  int idx = blockIdx.x * 256 + threadIdx.x;  // BH*MP*1024
  int t = idx & 1023;
  int m = (idx >> 10) % MP;
  int bh = idx / (MP * 1024);
  float val = 0.f;
  if (m < MF) {
    float pr = projk[((size_t)bh * 1024 + t) * MP + m];
    val = expf(pr - diagk[(size_t)bh * 1024 + t] - stab[bh]) * INV_SQRT_M + 1e-6f;
  }
  kft[idx] = (_Float16)val;
}

__global__ __launch_bounds__(256) void k_kfsum(const _Float16* __restrict__ kft,
                                               float* __restrict__ ksum) {
  int wid = (blockIdx.x * 256 + threadIdx.x) >> 5, lane = threadIdx.x & 31;  // BH*MP waves
  const _Float16* p = kft + (size_t)wid * 1024;
  float s = 0.f;
  for (int t = lane; t < 1024; t += 32) s += (float)p[t];
  s = wsum(s);
  if (lane == 0) ksum[wid] = s;
}

// kv^T = v3^T @ kf : [192,1024]x[1024,288] per head, K=1024
__global__ __launch_bounds__(256) void k_gemm_kvT(const _Float16* __restrict__ v3t,
                                                  const _Float16* __restrict__ kft,
                                                  _Float16* __restrict__ kvt) {
  int gw = (blockIdx.x * 256 + threadIdx.x) >> 5, lane = threadIdx.x & 31;
  const int WPB = 12 * 6;
  int bh = gw / WPB, r = gw % WPB;
  int m0 = (r / 6) * 16, n0 = (r % 6) * 48;
  const _Float16* A = v3t + (size_t)bh * D3 * Tc + (size_t)m0 * Tc;
  const _Float16* Bt = kft + (size_t)bh * MP * Tc + (size_t)n0 * Tc;
  v8f acc[3] = {};
  gemm_core<3>(A, Tc, Bt, Tc, Tc, lane, acc);
  _Float16* Dp = kvt + (size_t)bh * D3 * MP;
  int rr = (lane >> 4) << 3, c0 = lane & 15;
#pragma unroll
  for (int i = 0; i < 3; ++i)
#pragma unroll
    for (int r8 = 0; r8 < 8; ++r8)
      Dp[(size_t)(m0 + rr + r8) * MP + (n0 + i * 16 + c0)] = (_Float16)acc[i][r8];
}

__global__ __launch_bounds__(256) void k_z(const _Float16* __restrict__ qf,
                                           const float* __restrict__ ksum,
                                           float* __restrict__ zinv) {
  int row = (blockIdx.x * 256 + threadIdx.x) >> 5, lane = threadIdx.x & 31;
  int bh = row >> 10;
  const _Float16* qp = qf + (size_t)row * MP;
  const float* kp = ksum + (size_t)bh * MP;
  float s = 0.f;
  for (int m = lane; m < MP; m += 32) s += (float)qp[m] * kp[m];
  s = wsum(s);
  if (lane == 0) zinv[row] = 1.f / (s + 1e-6f);
}

// t_perf = (qf @ kv) * zinv : [1024,288]x[288,192] per head
__global__ __launch_bounds__(256) void k_gemm_tperf(const _Float16* __restrict__ qf,
                                                    const _Float16* __restrict__ kvt,
                                                    const float* __restrict__ zinv,
                                                    _Float16* __restrict__ tperf) {
  int gw = (blockIdx.x * 256 + threadIdx.x) >> 5, lane = threadIdx.x & 31;
  const int WPB = 64 * 4;
  int bh = gw / WPB, r = gw % WPB;
  int m0 = (r / 4) * 16, n0 = (r % 4) * 48;
  const _Float16* A = qf + (size_t)bh * Tc * MP + (size_t)m0 * MP;
  const _Float16* Bt = kvt + (size_t)bh * D3 * MP + (size_t)n0 * MP;
  v8f acc[3] = {};
  gemm_core<3>(A, MP, Bt, MP, MP, lane, acc);
  _Float16* Dp = tperf + (size_t)bh * Tc * D3;
  const float* zp = zinv + (size_t)bh * Tc;
  int rr = (lane >> 4) << 3, c0 = lane & 15;
#pragma unroll
  for (int i = 0; i < 3; ++i)
#pragma unroll
    for (int r8 = 0; r8 < 8; ++r8) {
      int row = m0 + rr + r8;
      Dp[(size_t)row * D3 + (n0 + i * 16 + c0)] = (_Float16)(acc[i][r8] * zp[row]);
    }
}

// enc_raw = t_perf @ W_enc + b_enc : [1024,192]x[192,128]
__global__ __launch_bounds__(256) void k_gemm_enc(const _Float16* __restrict__ tperf,
                                                  const _Float16* __restrict__ wencT,
                                                  const float* __restrict__ benc,
                                                  float* __restrict__ enc) {
  int gw = (blockIdx.x * 256 + threadIdx.x) >> 5, lane = threadIdx.x & 31;
  const int WPB = 64 * 2;
  int bh = gw / WPB, r = gw % WPB;
  int m0 = (r / 2) * 16, n0 = (r % 2) * 64;
  const _Float16* A = tperf + (size_t)bh * Tc * D3 + (size_t)m0 * D3;
  const _Float16* Bt = wencT + (size_t)n0 * D3;
  v8f acc[4] = {};
  gemm_core<4>(A, D3, Bt, D3, D3, lane, acc);
  float* Dp = enc + (size_t)bh * Tc * D2;
  int rr = (lane >> 4) << 3, c0 = lane & 15;
#pragma unroll
  for (int i = 0; i < 4; ++i)
#pragma unroll
    for (int r8 = 0; r8 < 8; ++r8) {
      int col = n0 + i * 16 + c0;
      Dp[(size_t)(m0 + rr + r8) * D2 + col] = acc[i][r8] + benc[col];
    }
}

// LN1 + exact GELU -> h1 (f16); fused per-row scaler sigmoid(h1 . W_s[:,0] + b_s[0])
__global__ __launch_bounds__(256) void k_ln1_gelu_scale(
    const float* __restrict__ enc, const float* __restrict__ g, const float* __restrict__ be,
    const float* __restrict__ Ws, const float* __restrict__ bs,
    _Float16* __restrict__ h1, float* __restrict__ scalebuf) {
  int row = (blockIdx.x * 256 + threadIdx.x) >> 5, lane = threadIdx.x & 31;
  const float* er = enc + (size_t)row * D2;
  float x[4], s = 0.f;
#pragma unroll
  for (int r = 0; r < 4; ++r) { x[r] = er[lane + 32 * r]; s += x[r]; }
  s = wsum(s);
  float mu = s * (1.f / 128.f);
  float qv = 0.f;
#pragma unroll
  for (int r = 0; r < 4; ++r) { float d = x[r] - mu; qv += d * d; }
  qv = wsum(qv);
  float rs = rsqrtf(qv * (1.f / 128.f) + 1e-5f);
  float dot = 0.f;
#pragma unroll
  for (int r = 0; r < 4; ++r) {
    int c = lane + 32 * r;
    float ln = (x[r] - mu) * rs * g[c] + be[c];
    float ge = 0.5f * ln * (1.f + erff(ln * 0.70710678118f));
    h1[(size_t)row * D2 + c] = (_Float16)ge;
    dot += ge * Ws[2 * c];
  }
  dot = wsum(dot) + bs[0];
  if (lane == 0) scalebuf[row] = 1.f / (1.f + expf(-dot));
}

// row = h1 @ W_dec + b_dec, stored directly in ChannelSplit(4) layout rowc[b, h*4+split, t, x]
__global__ __launch_bounds__(256) void k_gemm_dec(const _Float16* __restrict__ h1,
                                                  const _Float16* __restrict__ wdecT,
                                                  const float* __restrict__ bdec,
                                                  _Float16* __restrict__ rowc) {
  int gw = (blockIdx.x * 256 + threadIdx.x) >> 5, lane = threadIdx.x & 31;
  const int WPB = 64 * 8;
  int bh = gw / WPB, r = gw % WPB;
  int m0 = (r / 8) * 16, n0 = (r % 8) * 64;
  const _Float16* A = h1 + (size_t)bh * Tc * D2 + (size_t)m0 * D2;
  const _Float16* Bt = wdecT + (size_t)n0 * D2;
  v8f acc[4] = {};
  gemm_core<4>(A, D2, Bt, D2, D2, lane, acc);
  int b = bh >> 4, h = bh & 15;
  int rr = (lane >> 4) << 3, c0 = lane & 15;
#pragma unroll
  for (int i = 0; i < 4; ++i)
#pragma unroll
    for (int r8 = 0; r8 < 8; ++r8) {
      int j = n0 + i * 16 + c0;
      int row = m0 + rr + r8;
      int c = h * 4 + (j >> 7);
      float val = acc[i][r8] + bdec[j];
      rowc[((size_t)(b * 64 + c) * Tc + row) * 128 + (j & 127)] = (_Float16)val;
    }
}

// ================= CNN stage (direct, L2-resident weights) =================
__global__ __launch_bounds__(256) void k_conv1(const _Float16* __restrict__ in,
                                               const float* __restrict__ w,
                                               const float* __restrict__ bias,
                                               _Float16* __restrict__ out) {
  int idx = blockIdx.x * 256 + threadIdx.x;            // [4,64,512,128]
  int x = idx & 127, y = (idx >> 7) & 511, co = (idx >> 16) & 63, b = idx >> 22;
  float acc = bias[co];
  int ybase = y * 2 - 1;
  for (int ci = 0; ci < 64; ++ci) {
    const _Float16* ib = in + (size_t)(b * 64 + ci) * 1024 * 128;
    const float* wb = w + (size_t)(co * 64 + ci) * 9;
#pragma unroll
    for (int ky = 0; ky < 3; ++ky) {
      int iy = ybase + ky;
      if ((unsigned)iy < 1024u) {
        const _Float16* irow = ib + (size_t)iy * 128;
#pragma unroll
        for (int kx = 0; kx < 3; ++kx) {
          int ix = x + kx - 1;
          if ((unsigned)ix < 128u) acc += (float)irow[ix] * wb[ky * 3 + kx];
        }
      }
    }
  }
  out[idx] = (_Float16)fmaxf(acc, 0.f);
}

__global__ __launch_bounds__(256) void k_conv2(const _Float16* __restrict__ in,
                                               const float* __restrict__ w,
                                               const float* __restrict__ bias,
                                               _Float16* __restrict__ out) {
  int idx = blockIdx.x * 256 + threadIdx.x;            // [4,64,512,128]
  int x = idx & 127, y = (idx >> 7) & 511, co = (idx >> 16) & 63, b = idx >> 22;
  float acc = bias[co];
  for (int ci = 0; ci < 64; ++ci) {
    const _Float16* ib = in + (size_t)(b * 64 + ci) * 512 * 128;
    const float* wb = w + (size_t)(co * 64 + ci) * 9;
#pragma unroll
    for (int ky = 0; ky < 3; ++ky) {
      int iy = y + ky - 1;
      if ((unsigned)iy < 512u) {
        const _Float16* irow = ib + (size_t)iy * 128;
#pragma unroll
        for (int kx = 0; kx < 3; ++kx) {
          int ix = x + kx - 1;
          if ((unsigned)ix < 128u) acc += (float)irow[ix] * wb[ky * 3 + kx];
        }
      }
    }
  }
  out[idx] = (_Float16)fmaxf(acc, 0.f);
}

// conv3 with implicit 2x repeat-upsample of y2 along height (u[yy] = y2[yy>>1])
__global__ __launch_bounds__(256) void k_conv3(const _Float16* __restrict__ in,
                                               const float* __restrict__ w,
                                               const float* __restrict__ bias,
                                               _Float16* __restrict__ out) {
  int idx = blockIdx.x * 256 + threadIdx.x;            // [4,16,1024,128]
  int x = idx & 127, y = (idx >> 7) & 1023, co = (idx >> 17) & 15, b = idx >> 21;
  float acc = bias[co];
  for (int ci = 0; ci < 64; ++ci) {
    const _Float16* ib = in + (size_t)(b * 64 + ci) * 512 * 128;
    const float* wb = w + (size_t)(co * 64 + ci) * 9;
#pragma unroll
    for (int ky = 0; ky < 3; ++ky) {
      int iy = y + ky - 1;
      if ((unsigned)iy < 1024u) {
        const _Float16* irow = ib + (size_t)(iy >> 1) * 128;
#pragma unroll
        for (int kx = 0; kx < 3; ++kx) {
          int ix = x + kx - 1;
          if ((unsigned)ix < 128u) acc += (float)irow[ix] * wb[ky * 3 + kx];
        }
      }
    }
  }
  out[idx] = (_Float16)acc;
}

// ========== linear resize 128->256, softmax, top-16 threshold (== T-domain top-64) ==========
__global__ __launch_bounds__(256) void k_probs(const _Float16* __restrict__ y3,
                                               float* __restrict__ probsL,
                                               float* __restrict__ thr) {
  __shared__ float yS[8][128];
  int wid = threadIdx.x >> 5, lane = threadIdx.x & 31;
  int row = blockIdx.x * 8 + wid;            // 65536 rows = [b,h,t]
  const _Float16* yr = y3 + (size_t)row * 128;
#pragma unroll
  for (int r = 0; r < 4; ++r) yS[wid][lane + 32 * r] = (float)yr[lane + 32 * r];
  __syncthreads();
  float e[8];
  float mx = -3.4e38f;
#pragma unroll
  for (int r = 0; r < 8; ++r) {
    int j = lane + 32 * r;
    int i = j >> 1;
    if ((j & 1) == 0) {
      float a = yS[wid][i > 0 ? i - 1 : 0];
      e[r] = 0.25f * a + 0.75f * yS[wid][i];
    } else {
      float bb = yS[wid][i < 127 ? i + 1 : 127];
      e[r] = 0.75f * yS[wid][i] + 0.25f * bb;
    }
    mx = fmaxf(mx, e[r]);
  }
  mx = wmaxr(mx);
  float s = 0.f;
#pragma unroll
  for (int r = 0; r < 8; ++r) { e[r] = expf(e[r] - mx); s += e[r]; }
  s = wsum(s);
  float inv = 1.f / s;
  float p[8];
#pragma unroll
  for (int r = 0; r < 8; ++r) {
    p[r] = e[r] * inv;
    probsL[(size_t)row * 256 + lane + 32 * r] = p[r];
  }
  // exact 16th-largest (with tie multiplicity) via iterative max extraction
  float cur = 3.4e38f, tv = -3.4e38f;
  int remaining = 16;
  for (int it = 0; it < 16 && remaining > 0; ++it) {
    float m = -3.4e38f;
#pragma unroll
    for (int r = 0; r < 8; ++r) if (p[r] < cur) m = fmaxf(m, p[r]);
    m = wmaxr(m);
    if (m < -3.0e38f) break;
    float c = 0.f;
#pragma unroll
    for (int r = 0; r < 8; ++r) if (p[r] == m) c += 1.f;
    c = wsum(c);
    tv = m;
    remaining -= (int)c;
    cur = m;
  }
  if (lane == 0) thr[row] = tv;
}

// ========== top-k sparse attention (gather), scores = q16.k16 == q.k/sqrt(D) ==========
__global__ __launch_bounds__(128) void k_sparse_attn(
    const _Float16* __restrict__ q16, const _Float16* __restrict__ k16,
    const float* __restrict__ v, const float* __restrict__ amask,
    const float* __restrict__ probsL, const float* __restrict__ thr,
    const float* __restrict__ scalebuf, float* __restrict__ ctx) {
  __shared__ float sS[4][256];
  __shared__ int   sI[4][256];
  int wid = threadIdx.x >> 5, lane = threadIdx.x & 31;
  int row = blockIdx.x * 4 + wid;            // [b,h,t]
  int bh = row >> 10, b = bh >> 4;
  const _Float16* qr = q16 + (size_t)row * 64;
  float q0 = (float)qr[lane], q1 = (float)qr[lane + 32];
  const _Float16* kb = k16 + (size_t)bh * 1024 * 64;
  const float* vb = v + (size_t)bh * 1024 * 64;
  const float* pl = probsL + (size_t)row * 256;
  const float* am = amask + (size_t)b * 1024;
  float th = thr[row];
  int count = 0;
  for (int j = 0; j < 256; ++j) {
    if (count > 252) break;
    if (pl[j] >= th) {
      int sbase = j << 2;
#pragma unroll
      for (int ss = 0; ss < 4; ++ss) {
        int sidx = sbase + ss;
        const _Float16* kr = kb + (size_t)sidx * 64;
        float part = q0 * (float)kr[lane] + q1 * (float)kr[lane + 32];
        float sc = wsum(part) + am[sidx];
        if (lane == 0) { sS[wid][count] = sc; sI[wid][count] = sidx; }
        ++count;
      }
    }
  }
  lds_fence();
  float mx = -3.4e38f;
  for (int i = lane; i < count; i += 32) mx = fmaxf(mx, sS[wid][i]);
  mx = wmaxr(mx);
  float ssum = 0.f;
  for (int i = lane; i < count; i += 32) {
    float e = expf(sS[wid][i] - mx);
    sS[wid][i] = e;
    ssum += e;
  }
  ssum = wsum(ssum);
  lds_fence();
  float inv = 1.f / ssum;   // masked-out (-1e9) terms vanish; matches softmax*mask
  float c0 = 0.f, c1 = 0.f;
  for (int i = 0; i < count; ++i) {
    float p = sS[wid][i] * inv;
    const float* vr = vb + (size_t)sI[wid][i] * 64;
    c0 += p * vr[lane];
    c1 += p * vr[lane + 32];
  }
  float sc = scalebuf[row];
  float* cr = ctx + (size_t)row * 64;
  cr[lane] = c0 * sc;
  cr[lane + 32] = c1 * sc;
}

// ========== final transpose + LayerNorm over H*D ==========
__global__ __launch_bounds__(256) void k_ln2_out(const float* __restrict__ ctx,
                                                 const float* __restrict__ g,
                                                 const float* __restrict__ be,
                                                 float* __restrict__ out) {
  __shared__ float red[8];
  __shared__ float bc[2];
  int b = blockIdx.x >> 10, t = blockIdx.x & 1023;
  int tid = threadIdx.x, lane = tid & 31, wid = tid >> 5;
  float x[4], s = 0.f;
#pragma unroll
  for (int r = 0; r < 4; ++r) {
    int c = tid + 256 * r;
    x[r] = ctx[((size_t)(b * 16 + (c >> 6)) * 1024 + t) * 64 + (c & 63)];
    s += x[r];
  }
  s = wsum(s);
  if (lane == 0) red[wid] = s;
  __syncthreads();
  if (tid == 0) { float a = 0.f; for (int i = 0; i < 8; ++i) a += red[i]; bc[0] = a * (1.f / 1024.f); }
  __syncthreads();
  float mu = bc[0];
  float qv = 0.f;
#pragma unroll
  for (int r = 0; r < 4; ++r) { float d = x[r] - mu; qv += d * d; }
  qv = wsum(qv);
  __syncthreads();
  if (lane == 0) red[wid] = qv;
  __syncthreads();
  if (tid == 0) { float a = 0.f; for (int i = 0; i < 8; ++i) a += red[i]; bc[1] = rsqrtf(a * (1.f / 1024.f) + 1e-5f); }
  __syncthreads();
  float rs = bc[1];
#pragma unroll
  for (int r = 0; r < 4; ++r) {
    int c = tid + 256 * r;
    out[((size_t)b * 1024 + t) * 1024 + c] = (x[r] - mu) * rs * g[c] + be[c];
  }
}

}  // namespace pk

extern "C" void kernel_launch(void* const* d_in, const int* in_sizes, int n_in,
                              void* d_out, int out_size, void* d_ws, size_t ws_size,
                              hipStream_t stream) {
  using namespace pk;
  (void)in_sizes; (void)n_in; (void)out_size;
  if (ws_size < WS_NEED) return;

  const float* q     = (const float*)d_in[0];
  const float* k     = (const float*)d_in[1];
  const float* v     = (const float*)d_in[2];
  const float* amask = (const float*)d_in[3];
  const float* Wperf = (const float*)d_in[4];
  const float* Wenc  = (const float*)d_in[5];
  const float* benc  = (const float*)d_in[6];
  const float* ln1g  = (const float*)d_in[7];
  const float* ln1b  = (const float*)d_in[8];
  const float* Wdec  = (const float*)d_in[9];
  const float* bdec  = (const float*)d_in[10];
  const float* c1w   = (const float*)d_in[11];
  const float* c1b   = (const float*)d_in[12];
  const float* c2w   = (const float*)d_in[13];
  const float* c2b   = (const float*)d_in[14];
  const float* c3w   = (const float*)d_in[15];
  const float* c3b   = (const float*)d_in[16];
  const float* Ws    = (const float*)d_in[17];
  const float* bs    = (const float*)d_in[18];
  const float* ln2g  = (const float*)d_in[19];
  const float* ln2b  = (const float*)d_in[20];

  char* ws = (char*)d_ws;
  _Float16* q16   = (_Float16*)(ws + OFF_Q16);
  _Float16* k16   = (_Float16*)(ws + OFF_K16);
  float* diagq    = (float*)(ws + OFF_DIAGQ);
  float* diagk    = (float*)(ws + OFF_DIAGK);
  float* zinv     = (float*)(ws + OFF_ZINV);
  float* scalebuf = (float*)(ws + OFF_SCALE);
  float* thr      = (float*)(ws + OFF_THR);
  float* krmax    = (float*)(ws + OFF_KRMAX);
  float* kstab    = (float*)(ws + OFF_KSTAB);
  float* kfsum    = (float*)(ws + OFF_KFSUM);
  _Float16* wp16  = (_Float16*)(ws + OFF_WP16);
  _Float16* wencT = (_Float16*)(ws + OFF_WENCT);
  _Float16* wdecT = (_Float16*)(ws + OFF_WDECT);
  _Float16* v3t   = (_Float16*)(ws + OFF_V3T);
  _Float16* qf    = (_Float16*)(ws + OFF_QF);
  _Float16* kft   = (_Float16*)(ws + OFF_KFT);
  _Float16* kvt   = (_Float16*)(ws + OFF_KVT);
  _Float16* tperf = (_Float16*)(ws + OFF_TPERF);
  float* enc      = (float*)(ws + OFF_ENC);
  _Float16* h1    = (_Float16*)(ws + OFF_H1);
  _Float16* rowc  = (_Float16*)(ws + OFF_ROWC);
  float* projq    = (float*)(ws + OFF_PROJQ);
  float* projk    = (float*)(ws + OFF_PROJK);
  float* ctx      = (float*)(ws + OFF_CTX);
  _Float16* y1    = (_Float16*)(ws + OFF_Y1);
  _Float16* y2    = (_Float16*)(ws + OFF_Y2);
  _Float16* y3    = (_Float16*)(ws + OFF_Y3);
  float* probsL   = (float*)(ws + OFF_PROBS);
  float* outp     = (float*)d_out;

  k_prep_weights  <<<424,   256, 0, stream>>>(Wperf, Wenc, Wdec, wp16, wencT, wdecT);
  k_scale_convert <<<16384, 256, 0, stream>>>(q, k, q16, k16, diagq, diagk);
  k_build_v3T     <<<49152, 256, 0, stream>>>(v, v3t);
  k_gemm_proj     <<<3072,  256, 0, stream>>>(q16, wp16, projq);
  k_gemm_proj     <<<3072,  256, 0, stream>>>(k16, wp16, projk);
  k_qf            <<<8192,  256, 0, stream>>>(projq, diagq, qf);
  k_krowmax       <<<8192,  256, 0, stream>>>(projk, krmax);
  k_kstab         <<<8,     256, 0, stream>>>(krmax, kstab);
  k_kf            <<<73728, 256, 0, stream>>>(projk, diagk, kstab, kft);
  k_kfsum         <<<2304,  256, 0, stream>>>(kft, kfsum);
  k_gemm_kvT      <<<576,   256, 0, stream>>>(v3t, kft, kvt);
  k_z             <<<8192,  256, 0, stream>>>(qf, kfsum, zinv);
  k_gemm_tperf    <<<2048,  256, 0, stream>>>(qf, kvt, zinv, tperf);
  k_gemm_enc      <<<1024,  256, 0, stream>>>(tperf, wencT, benc, enc);
  k_ln1_gelu_scale<<<8192,  256, 0, stream>>>(enc, ln1g, ln1b, Ws, bs, h1, scalebuf);
  k_gemm_dec      <<<4096,  256, 0, stream>>>(h1, wdecT, bdec, rowc);
  k_conv1         <<<65536, 256, 0, stream>>>(rowc, c1w, c1b, y1);
  k_conv2         <<<65536, 256, 0, stream>>>(y1, c2w, c2b, y2);
  k_conv3         <<<32768, 256, 0, stream>>>(y2, c3w, c3b, y3);
  k_probs         <<<8192,  256, 0, stream>>>(y3, probsL, thr);
  k_sparse_attn   <<<16384, 128, 0, stream>>>(q16, k16, v, amask, probsL, thr, scalebuf, ctx);
  k_ln2_out       <<<4096,  256, 0, stream>>>(ctx, ln2g, ln2b, outp);
}
